// HOConv2d_69853348102514
// MI455X (gfx1250) — matrix operand order repaired and checked by reading, not compiled, not run on hardware
//
#include <hip/hip_runtime.h>
#include <hip/hip_bf16.h>

// ---------------------------------------------------------------------------
// HOConv2d on MI455X (gfx1250): pooled->dynamic-kernel regressor + per-sample
// implicit-GEMM conv via v_wmma_f32_16x16x32_f16.
//
// Roofline: conv = 2.27 GFLOP vs ~48 MB HBM traffic -> memory bound (~2us at
// 23.3 TB/s). So: read x once (LDS slab per (n,oh), async-to-LDS), write out
// once, run MACs through WMMA with f32 accumulation. K = IC*3*3 = 144 padded
// to 160 -> 5 K-chunks of 32 per 16x16 tile; each wave computes a 32x16
// output block (2 accumulators, shared B fragment).
// ---------------------------------------------------------------------------

#define NB   64
#define IC   16
#define HH   64
#define WW   64
#define OC   32
#define OH   62
#define OW   62
#define KTOT 144      // IC*3*3
#define KPAD 160      // padded to 5 chunks of K=32

typedef __attribute__((ext_vector_type(16))) _Float16 v16h;
typedef __attribute__((ext_vector_type(8)))  _Float16 v8h;
typedef __attribute__((ext_vector_type(8)))  float    v8f;

// Exact pointee type the async-to-LDS builtin expects (per hipcc diagnostic):
// "__attribute__((__vector_size__(4 * sizeof(int)))) int __device__ *"
typedef int vec4i __attribute__((vector_size(4 * sizeof(int))));

// Address-space casts (generic LDS pointers carry the raw LDS offset in the
// low 32 bits on AMDGPU, so the integer truncation is the standard trick).
#define AS1_I4(p) ((__attribute__((address_space(1))) vec4i*)(unsigned long long)(p))
#define AS3_I4(p) ((__attribute__((address_space(3))) vec4i*)(unsigned int)(unsigned long long)(p))

#if __has_builtin(__builtin_amdgcn_global_load_async_to_lds_b128)
#define USE_ASYNC_LDS 1
#else
#define USE_ASYNC_LDS 0
#endif

__device__ __forceinline__ void wait_async0() {
#if __has_builtin(__builtin_amdgcn_s_wait_asynccnt)
  __builtin_amdgcn_s_wait_asynccnt(0);
#else
  asm volatile("s_wait_asynccnt 0" ::: "memory");
#endif
}

// ---------------------------------------------------------------------------
// Stage 1: pooled[n][c] = mean over HxW of x[n][c]
// ---------------------------------------------------------------------------
__global__ __launch_bounds__(256) void pool_kernel(const float* __restrict__ x,
                                                   float* __restrict__ pooled) {
  __shared__ float red[256];
  const int n = blockIdx.x, tid = threadIdx.x;
  for (int c = 0; c < IC; ++c) {
    float s = 0.f;
    const float* p = x + ((size_t)n * IC + c) * (HH * WW);
    for (int i = tid; i < HH * WW; i += 256) s += p[i];
    red[tid] = s;
    __syncthreads();
    for (int st = 128; st > 0; st >>= 1) {
      if (tid < st) red[tid] += red[tid + st];
      __syncthreads();
    }
    if (tid == 0) pooled[n * IC + c] = red[0] * (1.0f / (HH * WW));
    __syncthreads();
  }
}

// ---------------------------------------------------------------------------
// Stage 2: kerh[n][oc][kpad] = f16( pooled[n] . W_reg[:,kflat] + b_reg[kflat] )
// kflat = oc*144 + (ic*9 + kh*3 + kw); k in [144,160) zero-padded.
// ---------------------------------------------------------------------------
__global__ __launch_bounds__(256) void reg_kernel(const float* __restrict__ pooled,
                                                  const float* __restrict__ Wreg,
                                                  const float* __restrict__ breg,
                                                  _Float16* __restrict__ kerh) {
  const int idx = blockIdx.x * 256 + threadIdx.x;  // [0, 64*32*160)
  const int kl  = idx % KPAD;
  const int oc  = (idx / KPAD) & (OC - 1);
  const int n   = idx / (KPAD * OC);
  float v = 0.f;
  if (kl < KTOT) {
    const int kf = oc * KTOT + kl;
    v = breg[kf];
#pragma unroll
    for (int ic = 0; ic < IC; ++ic)
      v += pooled[n * IC + ic] * Wreg[ic * (OC * KTOT) + kf];
  }
  kerh[idx] = (_Float16)v;
}

// ---------------------------------------------------------------------------
// Stage 3: implicit-GEMM conv. grid = N*OH blocks, 128 threads (4 waves).
// Block (n, oh): stage x[n, 0:16, oh..oh+2, 0:64] (12 KB) into LDS; wave w
// computes out[n, 0:32, oh, w*16 .. w*16+15] with 2 WMMA accumulators.
// ---------------------------------------------------------------------------
__global__ __launch_bounds__(128) void conv_iwk_wmma(const float* __restrict__ x,
                                                     const _Float16* __restrict__ kerh,
                                                     float* __restrict__ out) {
  __shared__ float xs[IC * 3 * WW + 68];  // 3072 slab + zero pad
  __shared__ int   koff[KPAD];

  const int b    = blockIdx.x;
  const int n    = b / OH;
  const int oh   = b % OH;
  const int tid  = threadIdx.x;
  const int lane = tid & 31;
  const int wave = tid >> 5;

  // ---- stage the input slab: 768 chunks of 16B, 6 per thread ----
  for (int i = tid; i < (IC * 3 * WW) / 4; i += 128) {
    const int rr = i >> 4;          // ic*3 + r
    const int c4 = (i & 15) << 2;   // starting column
    const int ic = rr / 3;
    const int r  = rr - 3 * ic;
    const float* src = x + (((size_t)n * IC + ic) * HH + (oh + r)) * WW + c4;
#if USE_ASYNC_LDS
    __builtin_amdgcn_global_load_async_to_lds_b128(AS1_I4(src), AS3_I4(&xs[i * 4]), 0, 0);
#else
    *(float4*)(&xs[i * 4]) = *(const float4*)src;
#endif
  }
  // k -> LDS offset table; invalid k points into the zeroed pad region.
  if (tid < KPAD) {
    int o = IC * 3 * WW;
    if (tid < KTOT) {
      const int ic = tid / 9, rem = tid % 9;
      o = (ic * 3 + rem / 3) * WW + (rem % 3);
    }
    koff[tid] = o;
  }
  if (tid < 68) xs[IC * 3 * WW + tid] = 0.0f;
#if USE_ASYNC_LDS
  wait_async0();
#endif
  __syncthreads();

  // ---- per-lane fragment coordinates (ISA 16x16 WMMA layouts) ----
  const int nn   = lane & 15;   // A row (oc%16), B/D column
  const int hs   = lane >> 4;   // half-wave select
  const int ow   = wave * 16 + nn;
  const int hs16 = hs << 4;     // B K-base offset
  const int h8   = hs << 3;     // D row offset

  const _Float16* A0 = kerh + ((size_t)n * OC + nn) * KPAD;   // oc = nn
  const _Float16* A1 = A0 + (size_t)16 * KPAD;                // oc = 16+nn

  v8f acc0 = {};
  v8f acc1 = {};

#pragma unroll
  for (int kk = 0; kk < KPAD / 32; ++kk) {
    const int kb = kk * 32;

    // B fragment: lane holds column ow, K = kb + hs16 + {0..15}
    v16h bfrag;
#pragma unroll
    for (int v = 0; v < 8; ++v) {
      const int k0 = kb + hs16 + 2 * v;
      bfrag[2 * v]     = (_Float16)xs[koff[k0] + ow];
      bfrag[2 * v + 1] = (_Float16)xs[koff[k0 + 1] + ow];
    }

    // A fragments: VGPR0-3 hold K = kb+hs16+{0..7}, VGPR4-7 K = kb+hs16+8+{0..7}
    v8h al0 = *(const v8h*)(A0 + kb + hs16);
    v8h ah0 = *(const v8h*)(A0 + kb + hs16 + 8);
    v8h al1 = *(const v8h*)(A1 + kb + hs16);
    v8h ah1 = *(const v8h*)(A1 + kb + hs16 + 8);
    v16h a0, a1;
#pragma unroll
    for (int i = 0; i < 8; ++i) {
      a0[i] = al0[i]; a0[i + 8] = ah0[i];
      a1[i] = al1[i]; a1[i + 8] = ah1[i];
    }

    acc0 = __builtin_amdgcn_wmma_f32_16x16x32_f16(false, a0, false, bfrag,
                                                  (short)0, acc0, false, false);
    acc1 = __builtin_amdgcn_wmma_f32_16x16x32_f16(false, a1, false, bfrag,
                                                  (short)0, acc1, false, false);
  }

  // ---- store D: VGPR r -> row (h8 + r); columns ow >= 62 are padding ----
  if (ow < OW) {
    const size_t base = (((size_t)n * OC + h8) * OH + oh) * OW + ow;
#pragma unroll
    for (int r = 0; r < 8; ++r) {
      out[base + (size_t)r * (OH * OW)]        = acc0[r];   // oc = h8 + r
      out[base + (size_t)(r + 16) * (OH * OW)] = acc1[r];   // oc = 16 + h8 + r
    }
  }
}

// ---------------------------------------------------------------------------
// Host-side launcher
// ---------------------------------------------------------------------------
extern "C" void kernel_launch(void* const* d_in, const int* in_sizes, int n_in,
                              void* d_out, int out_size, void* d_ws, size_t ws_size,
                              hipStream_t stream) {
  (void)in_sizes; (void)n_in; (void)out_size; (void)ws_size;
  const float* x    = (const float*)d_in[0];   // [64,16,64,64]
  const float* Wreg = (const float*)d_in[1];   // [16,4608]
  const float* breg = (const float*)d_in[2];   // [4608]
  float* out = (float*)d_out;                  // [64,32,62,62]

  float*    pooled = (float*)d_ws;                       // 64*16 f32 (4 KB)
  _Float16* kerh   = (_Float16*)((char*)d_ws + 4096);    // 64*32*160 f16 (640 KB)

  pool_kernel<<<NB, 256, 0, stream>>>(x, pooled);
  reg_kernel<<<(NB * OC * KPAD) / 256, 256, 0, stream>>>(pooled, Wreg, breg, kerh);
  conv_iwk_wmma<<<NB * OH, 128, 0, stream>>>(x, kerh, out);
}
